// get_model_35433480192231
// MI455X (gfx1250) — matrix operand order repaired and checked by reading, not compiled, not run on hardware
//
#include <hip/hip_runtime.h>
#include <hip/hip_bf16.h>
#include <stdint.h>

// ---------------------------------------------------------------------------
// CDNA5 (gfx1250) wave32 WMMA helpers
// ---------------------------------------------------------------------------
typedef __bf16 bf16_t;
typedef __attribute__((ext_vector_type(16))) __bf16 v16bf;
typedef __attribute__((ext_vector_type(8)))  __bf16 v8bf;
typedef __attribute__((ext_vector_type(8)))  float  v8f;

#define BATCH 16
#define NPTS  4096

__device__ __forceinline__ v8f wmma_bf16(v16bf a, v16bf b, v8f c) {
  // v_wmma_f32_16x16x32_bf16  D = A(16x32) * B(32x16) + C(16x16 f32)
  return __builtin_amdgcn_wmma_f32_16x16x32_bf16(false, a, false, b,
                                                 (short)0, c, false, false);
}

// A fragment (16x32 bf16) built from a row-major f32 matrix in global memory.
// Lane L (0..15): row = row0+L, K run [k0 + 0..7] and [k0+16..23]
// Lane L (16..31): row = row0+(L-15? no: L-16), K run [k0+8..15],[k0+24..31]
__device__ __forceinline__ v16bf afrag_f32(const float* A, int lda, int row0, int k0) {
  const int lane = threadIdx.x & 31;
  const float* p = A + (size_t)(row0 + (lane & 15)) * lda + k0 + ((lane >> 4) << 3);
  v16bf a;
#pragma unroll
  for (int i = 0; i < 8; ++i) {
    a[i]     = (bf16_t)p[i];
    a[8 + i] = (bf16_t)p[16 + i];
  }
  return a;
}

// A fragment from a row-major bf16 tile staged in LDS (two ds_load_b128).
__device__ __forceinline__ v16bf afrag_lds(const bf16_t* As, int lda, int row0, int k0) {
  const int lane = threadIdx.x & 31;
  const bf16_t* p = As + (row0 + (lane & 15)) * lda + k0 + ((lane >> 4) << 3);
  v8bf lo = *(const v8bf*)p;
  v8bf hi = *(const v8bf*)(p + 16);
  v16bf a;
#pragma unroll
  for (int i = 0; i < 8; ++i) { a[i] = lo[i]; a[8 + i] = hi[i]; }
  return a;
}

// B fragment: weights pre-swizzled so each lane's 16 bf16 values are a single
// contiguous 32-byte chunk:  offset = ((nt*KC + kc)*32 + lane)*16
__device__ __forceinline__ v16bf bfrag(const bf16_t* Bsw, int nt, int kc, int KC) {
  const bf16_t* p = Bsw + (((size_t)(nt * KC + kc)) << 9) + ((threadIdx.x & 31) << 4);
  return *(const v16bf*)p;
}

// ---------------------------------------------------------------------------
// Weight pack: f32 (Kin x Nout) row-major -> bf16 WMMA B-fragment order,
// zero-padded to K multiple of 32 / N multiple of 16.
// element e of lane: K = kc*32 + (lane>>4)*8 + (e&7) + (e>>3)*16 ; N = nt*16 + (lane&15)
// ---------------------------------------------------------------------------
__global__ void k_pack_w(const float* __restrict__ W, bf16_t* __restrict__ out,
                         int Kin, int Nout, int KC, int NT) {
  int tid = blockIdx.x * 256 + threadIdx.x;
  int total = NT * KC * 512;
  if (tid >= total) return;
  int e    = tid & 15;
  int lane = (tid >> 4) & 31;
  int t3   = tid >> 9;
  int kc   = t3 % KC;
  int nt   = t3 / KC;
  int K  = kc * 32 + ((lane >> 4) << 3) + (e & 7) + ((e >> 3) << 4);
  int Nc = nt * 16 + (lane & 15);
  float v = (K < Kin && Nc < Nout) ? W[(size_t)K * Nout + Nc] : 0.0f;
  out[tid] = (bf16_t)v;
}

// ---------------------------------------------------------------------------
// xyz (B,3,N) -> xyzT (B,N,3)
// ---------------------------------------------------------------------------
__global__ void k_transpose(const float* __restrict__ xyz, float* __restrict__ xyzT) {
  int g = blockIdx.x * 256 + threadIdx.x;
  int total = BATCH * NPTS * 3;
  if (g >= total) return;
  int b = g / (NPTS * 3);
  int r = g % (NPTS * 3);
  int i = r / 3, c = r % 3;
  xyzT[g] = xyz[(size_t)b * 3 * NPTS + (size_t)c * NPTS + i];
}

// ---------------------------------------------------------------------------
// Farthest point sampling: one block per batch, LDS-resident distance array,
// 'npoint' sequential block-wide argmax steps (the true latency path).
// Matches lax.scan semantics: idx[t] = far BEFORE the t-th distance update.
// ---------------------------------------------------------------------------
__global__ void k_fps(const float* __restrict__ pts, int n, int npoint,
                      int* __restrict__ idx, float* __restrict__ new_xyz) {
  __shared__ float dist[4096];
  __shared__ float rv[256];
  __shared__ int   ri[256];
  __shared__ int   s_far;
  int b = blockIdx.x, tid = threadIdx.x;
  const float* P = pts + (size_t)b * n * 3;
  if (tid == 0) s_far = 0;
  for (int i = tid; i < n; i += 256) dist[i] = 1e10f;
  __syncthreads();
  for (int t = 0; t < npoint; ++t) {
    int far = s_far;
    float fx = P[far * 3 + 0], fy = P[far * 3 + 1], fz = P[far * 3 + 2];
    if (tid == 0) {
      idx[b * npoint + t] = far;
      new_xyz[((size_t)b * npoint + t) * 3 + 0] = fx;
      new_xyz[((size_t)b * npoint + t) * 3 + 1] = fy;
      new_xyz[((size_t)b * npoint + t) * 3 + 2] = fz;
    }
    float bv = -1.0f; int bi = 0;
    for (int i = tid; i < n; i += 256) {
      float dx = P[i * 3 + 0] - fx;
      float dy = P[i * 3 + 1] - fy;
      float dz = P[i * 3 + 2] - fz;
      float d  = dx * dx + dy * dy + dz * dz;
      float nd = fminf(dist[i], d);
      dist[i] = nd;
      if (nd > bv) { bv = nd; bi = i; }
    }
    rv[tid] = bv; ri[tid] = bi;
    __syncthreads();
    for (int s = 128; s > 0; s >>= 1) {
      if (tid < s) {
        if (rv[tid + s] > rv[tid] ||
            (rv[tid + s] == rv[tid] && ri[tid + s] < ri[tid])) {
          rv[tid] = rv[tid + s]; ri[tid] = ri[tid + s];
        }
      }
      __syncthreads();
    }
    if (tid == 0) s_far = ri[0];
    __syncthreads();
  }
}

// ---------------------------------------------------------------------------
// Ball query: ascending-index scan == sort()[:,:nsample] of the reference.
// ---------------------------------------------------------------------------
__global__ void k_ballquery(const float* __restrict__ xyz, const float* __restrict__ nxyz,
                            int n, int s, int nsample, float r2,
                            int* __restrict__ idx, int total) {
  int g = blockIdx.x * 256 + threadIdx.x;
  if (g >= total) return;
  int b = g / s;
  const float* P = xyz + (size_t)b * n * 3;
  float qx = nxyz[(size_t)g * 3 + 0];
  float qy = nxyz[(size_t)g * 3 + 1];
  float qz = nxyz[(size_t)g * 3 + 2];
  int* o = idx + (size_t)g * nsample;
  int cnt = 0, first = n - 1;
  for (int i = 0; i < n && cnt < nsample; ++i) {
    float dx = P[i * 3 + 0] - qx;
    float dy = P[i * 3 + 1] - qy;
    float dz = P[i * 3 + 2] - qz;
    float d = dx * dx + dy * dy + dz * dz;
    if (d <= r2) { if (cnt == 0) first = i; o[cnt++] = i; }
  }
  for (int k = cnt; k < nsample; ++k) o[k] = first;
}

// ---------------------------------------------------------------------------
// SA1 fused: group (32x6) -> WMMA (K padded to 32) -> relu+bias -> max-pool.
// grid (512, B), block 256 = 8 waves; wave w = N-tile w (N=128).
// ---------------------------------------------------------------------------
__global__ void k_sa1(const float* __restrict__ xyzT, const float* __restrict__ nx1,
                      const int* __restrict__ bq1, const bf16_t* __restrict__ Wsw,
                      const float* __restrict__ bias, float* __restrict__ l1) {
  __shared__ bf16_t As[32 * 32];
  int s = blockIdx.x, b = blockIdx.y, tid = threadIdx.x;
  const int*   gi = bq1 + ((size_t)(b * 512 + s)) * 32;
  const float* P  = xyzT + (size_t)b * NPTS * 3;
  const float* cx = nx1 + ((size_t)(b * 512 + s)) * 3;
  for (int e = tid; e < 32 * 32; e += 256) {
    int row = e >> 5, c = e & 31;
    float v = 0.0f;
    int pi = gi[row];
    if (c < 3)      v = P[pi * 3 + c] - cx[c];
    else if (c < 6) v = P[pi * 3 + (c - 3)];
    As[e] = (bf16_t)v;
  }
  __syncthreads();
  int wave = tid >> 5, lane = tid & 31;
  v16bf bf = bfrag(Wsw, wave, 0, 1);
  v8f acc0 = {}, acc1 = {};
  acc0 = wmma_bf16(afrag_lds(As, 32, 0, 0), bf, acc0);
  acc1 = wmma_bf16(afrag_lds(As, 32, 16, 0), bf, acc1);
  float m = -1e30f;
#pragma unroll
  for (int r = 0; r < 8; ++r) m = fmaxf(m, fmaxf(acc0[r], acc1[r]));
  m = fmaxf(m, __shfl_xor(m, 16, 32));       // merge the two row halves
  if (lane < 16) {
    int col = wave * 16 + lane;
    l1[((size_t)(b * 512 + s)) * 128 + col] = fmaxf(m + bias[col], 0.0f);
  }
}

// ---------------------------------------------------------------------------
// SA2 fused: group (64x131 -> LDS padded 64x160) -> 4 mtiles x 5 kchunks WMMA
// -> relu+bias -> max-pool.  grid (128, B), block 512 = 16 waves (= N-tiles).
// ---------------------------------------------------------------------------
__global__ void k_sa2(const float* __restrict__ l1xyz, const float* __restrict__ l1,
                      const float* __restrict__ nx2, const int* __restrict__ bq2,
                      const bf16_t* __restrict__ Wsw, const float* __restrict__ bias,
                      float* __restrict__ l2) {
  __shared__ bf16_t As[64 * 160];
  int s = blockIdx.x, b = blockIdx.y, tid = threadIdx.x;
  const int*   gi = bq2 + ((size_t)(b * 128 + s)) * 64;
  const float* PX = l1xyz + (size_t)b * 512 * 3;
  const float* PF = l1 + (size_t)b * 512 * 128;
  const float* cx = nx2 + ((size_t)(b * 128 + s)) * 3;
  for (int e = tid; e < 64 * 160; e += 512) {
    int row = e / 160, c = e % 160;
    float v = 0.0f;
    int pi = gi[row];
    if (c < 3)        v = PX[pi * 3 + c] - cx[c];
    else if (c < 131) v = PF[(size_t)pi * 128 + (c - 3)];
    As[e] = (bf16_t)v;
  }
  __syncthreads();
  int wave = tid >> 5, lane = tid & 31;
  float m = -1e30f;
  for (int mt = 0; mt < 4; ++mt) {
    v8f acc = {};
#pragma unroll
    for (int kc = 0; kc < 5; ++kc)
      acc = wmma_bf16(afrag_lds(As, 160, mt * 16, kc * 32),
                      bfrag(Wsw, wave, kc, 5), acc);
#pragma unroll
    for (int r = 0; r < 8; ++r) m = fmaxf(m, acc[r]);
  }
  m = fmaxf(m, __shfl_xor(m, 16, 32));
  if (lane < 16) {
    int col = wave * 16 + lane;
    l2[((size_t)(b * 128 + s)) * 256 + col] = fmaxf(m + bias[col], 0.0f);
  }
}

// ---------------------------------------------------------------------------
// SA3 prep: A3(B,128,288) = [l2_xyz | l2 | 0-pad]
// ---------------------------------------------------------------------------
__global__ void k_sa3prep(const float* __restrict__ nx2, const float* __restrict__ l2,
                          float* __restrict__ A3) {
  int g = blockIdx.x * 256 + threadIdx.x;
  int total = BATCH * 128 * 288;
  if (g >= total) return;
  int b = g / (128 * 288);
  int r = g % (128 * 288);
  int row = r / 288, c = r % 288;
  float v = 0.0f;
  if (c < 3)        v = nx2[((size_t)(b * 128 + row)) * 3 + c];
  else if (c < 259) v = l2[((size_t)(b * 128 + row)) * 256 + (c - 3)];
  A3[g] = v;
}

// ---------------------------------------------------------------------------
// SA3: per (ntile,b) block, 8 waves = 8 M-tiles, 9 k-chunks, then max over
// all 128 rows (in-wave + LDS across waves); writes l3 to ws and to d_out.
// ---------------------------------------------------------------------------
__global__ void k_sa3(const float* __restrict__ A3, const bf16_t* __restrict__ Wsw,
                      const float* __restrict__ bias, float* __restrict__ l3ws,
                      float* __restrict__ out_l3) {
  __shared__ float red[8][16];
  int nt = blockIdx.x, b = blockIdx.y, tid = threadIdx.x;
  int wave = tid >> 5, lane = tid & 31;
  const float* A = A3 + (size_t)b * 128 * 288;
  v8f acc = {};
#pragma unroll
  for (int kc = 0; kc < 9; ++kc)
    acc = wmma_bf16(afrag_f32(A, 288, wave * 16, kc * 32),
                    bfrag(Wsw, nt, kc, 9), acc);
  float m = -1e30f;
#pragma unroll
  for (int r = 0; r < 8; ++r) m = fmaxf(m, acc[r]);
  m = fmaxf(m, __shfl_xor(m, 16, 32));
  if (lane < 16) red[wave][lane] = m;
  __syncthreads();
  if (wave == 0 && lane < 16) {
    float v = red[0][lane];
#pragma unroll
    for (int w = 1; w < 8; ++w) v = fmaxf(v, red[w][lane]);
    int col = nt * 16 + lane;
    float r = fmaxf(v + bias[col], 0.0f);
    l3ws[(size_t)b * 1024 + col]  = r;
    out_l3[(size_t)b * 1024 + col] = r;
  }
}

// ---------------------------------------------------------------------------
// FP3 prep: A(B,128,1280) = [l2 (256) | l3 broadcast (1024)]
// ---------------------------------------------------------------------------
__global__ void k_fp3prep(const float* __restrict__ l2, const float* __restrict__ l3ws,
                          float* __restrict__ A) {
  int g = blockIdx.x * 256 + threadIdx.x;
  int total = BATCH * 128 * 1280;
  if (g >= total) return;
  int b = g / (128 * 1280);
  int r = g % (128 * 1280);
  int row = r / 1280, c = r % 1280;
  A[g] = (c < 256) ? l2[((size_t)(b * 128 + row)) * 256 + c]
                   : l3ws[(size_t)b * 1024 + (c - 256)];
}

// ---------------------------------------------------------------------------
// Generic batched GEMM: out = [relu](A @ Wsw + bias). One wave per 16x16
// output tile; A is f32 row-major with zero-padded K columns.
// grid = (ntiles, mtiles, batch), block = 32.
// ---------------------------------------------------------------------------
__global__ void k_gemm(const float* __restrict__ A, size_t aStride, int lda,
                       const bf16_t* __restrict__ Wsw, int KC,
                       const float* __restrict__ bias,
                       float* __restrict__ out, size_t oStride, int ldo,
                       int Nout, int do_relu) {
  int nt = blockIdx.x, mt = blockIdx.y, b = blockIdx.z;
  const float* Ab = A + (size_t)b * aStride;
  float* Ob = out + (size_t)b * oStride;
  v8f acc = {};
  for (int kc = 0; kc < KC; ++kc)
    acc = wmma_bf16(afrag_f32(Ab, lda, mt * 16, kc * 32),
                    bfrag(Wsw, nt, kc, KC), acc);
  int lane = threadIdx.x & 31;
  int col = nt * 16 + (lane & 15);
  if (col < Nout) {
    float bb = bias[col];
    int rbase = mt * 16 + ((lane >> 4) << 3);
#pragma unroll
    for (int r = 0; r < 8; ++r) {
      float v = acc[r] + bb;
      if (do_relu) v = fmaxf(v, 0.0f);
      Ob[(size_t)(rbase + r) * ldo + col] = v;
    }
  }
}

// ---------------------------------------------------------------------------
// 3-NN inverse-distance interpolation, writing into a cat buffer at col c0.
// ---------------------------------------------------------------------------
__global__ void k_knn(const float* __restrict__ xyz1, const float* __restrict__ xyz2,
                      const float* __restrict__ f2, int n1, int n2, int C2, int ld2,
                      float* __restrict__ cat, int ldcat, int c0, int total) {
  int g = blockIdx.x * 256 + threadIdx.x;
  if (g >= total) return;
  int b = g / n1, i = g % n1;
  const float* P1 = xyz1 + (size_t)b * n1 * 3;
  const float* P2 = xyz2 + (size_t)b * n2 * 3;
  float qx = P1[i * 3 + 0], qy = P1[i * 3 + 1], qz = P1[i * 3 + 2];
  float d0 = 1e30f, d1 = 1e30f, d2v = 1e30f;
  int i0 = 0, i1 = 0, i2 = 0;
  for (int j = 0; j < n2; ++j) {
    float dx = P2[j * 3 + 0] - qx;
    float dy = P2[j * 3 + 1] - qy;
    float dz = P2[j * 3 + 2] - qz;
    float d = dx * dx + dy * dy + dz * dz;
    if (d < d0)       { d2v = d1; i2 = i1; d1 = d0; i1 = i0; d0 = d; i0 = j; }
    else if (d < d1)  { d2v = d1; i2 = i1; d1 = d; i1 = j; }
    else if (d < d2v) { d2v = d; i2 = j; }
  }
  float w0 = 1.0f / (d0 + 1e-8f);
  float w1 = 1.0f / (d1 + 1e-8f);
  float w2 = 1.0f / (d2v + 1e-8f);
  float wsum = w0 + w1 + w2;
  w0 /= wsum; w1 /= wsum; w2 /= wsum;
  const float* F = f2 + (size_t)b * n2 * ld2;
  float* O = cat + ((size_t)(b * n1 + i)) * ldcat + c0;
  for (int c = 0; c < C2; ++c)
    O[c] = w0 * F[(size_t)i0 * ld2 + c] + w1 * F[(size_t)i1 * ld2 + c]
         + w2 * F[(size_t)i2 * ld2 + c];
}

// FP2 prep: cat2[:, 0:128] = l1
__global__ void k_fp2prep(const float* __restrict__ l1, float* __restrict__ cat2) {
  int g = blockIdx.x * 256 + threadIdx.x;
  int total = BATCH * 512 * 128;
  if (g >= total) return;
  int b = g / (512 * 128);
  int r = g % (512 * 128);
  int row = r / 128, c = r % 128;
  cat2[((size_t)(b * 512 + row)) * 384 + c] = l1[g];
}

// FP1 prep: cat1(B,4096,160): [cls(16) | l0_xyz(3) | pts0(3) | knn fills 22..149 | 0-pad]
__global__ void k_fp1prep(const float* __restrict__ cls, const float* __restrict__ xyzT,
                          float* __restrict__ cat1) {
  int g = blockIdx.x * 256 + threadIdx.x;
  int total = BATCH * NPTS;
  if (g >= total) return;
  int b = g / NPTS;
  float* O = cat1 + (size_t)g * 160;
  const float* cl = cls + (size_t)b * 16;
  for (int c = 0; c < 16; ++c) O[c] = cl[c];
  for (int c = 0; c < 3; ++c) {
    float v = xyzT[(size_t)g * 3 + c];
    O[16 + c] = v;
    O[19 + c] = v;
  }
  for (int c = 150; c < 160; ++c) O[c] = 0.0f;
}

__global__ void k_logsoftmax(const float* __restrict__ logits, float* __restrict__ out,
                             int total) {
  int g = blockIdx.x * 256 + threadIdx.x;
  if (g >= total) return;
  const float* L = logits + (size_t)g * 50;
  float m = -1e30f;
  for (int c = 0; c < 50; ++c) m = fmaxf(m, L[c]);
  float ssum = 0.0f;
  for (int c = 0; c < 50; ++c) ssum += __expf(L[c] - m);
  float lse = m + __logf(ssum);
  float* O = out + (size_t)g * 50;
  for (int c = 0; c < 50; ++c) O[c] = L[c] - lse;
}

// ---------------------------------------------------------------------------
// Host side
// ---------------------------------------------------------------------------
static inline size_t bump(size_t& off, size_t bytes) {
  size_t o = off;
  off = (off + bytes + 255) & ~(size_t)255;
  return o;
}

extern "C" void kernel_launch(void* const* d_in, const int* in_sizes, int n_in,
                              void* d_out, int out_size, void* d_ws, size_t ws_size,
                              hipStream_t stream) {
  (void)in_sizes; (void)n_in; (void)out_size; (void)ws_size;
  const float* xyz     = (const float*)d_in[0];
  const float* cls     = (const float*)d_in[1];
  const float* w_sa1   = (const float*)d_in[2];
  const float* b_sa1   = (const float*)d_in[3];
  const float* w_sa2   = (const float*)d_in[4];
  const float* b_sa2   = (const float*)d_in[5];
  const float* w_sa3   = (const float*)d_in[6];
  const float* b_sa3   = (const float*)d_in[7];
  const float* w_fp3   = (const float*)d_in[8];
  const float* b_fp3   = (const float*)d_in[9];
  const float* w_fp2   = (const float*)d_in[10];
  const float* b_fp2   = (const float*)d_in[11];
  const float* w_fp1   = (const float*)d_in[12];
  const float* b_fp1   = (const float*)d_in[13];
  const float* w_conv2 = (const float*)d_in[14];
  const float* b_conv2 = (const float*)d_in[15];
  float* out = (float*)d_out;

  char* ws = (char*)d_ws;
  size_t off = 0;
  float* xyzT  = (float*)(ws + bump(off, sizeof(float) * BATCH * NPTS * 3));
  int*   idx1  = (int*)  (ws + bump(off, sizeof(int)   * BATCH * 512));
  float* nx1   = (float*)(ws + bump(off, sizeof(float) * BATCH * 512 * 3));
  int*   bq1   = (int*)  (ws + bump(off, sizeof(int)   * BATCH * 512 * 32));
  float* l1    = (float*)(ws + bump(off, sizeof(float) * BATCH * 512 * 128));
  int*   idx2  = (int*)  (ws + bump(off, sizeof(int)   * BATCH * 128));
  float* nx2   = (float*)(ws + bump(off, sizeof(float) * BATCH * 128 * 3));
  int*   bq2   = (int*)  (ws + bump(off, sizeof(int)   * BATCH * 128 * 64));
  float* l2    = (float*)(ws + bump(off, sizeof(float) * BATCH * 128 * 256));
  float* A3    = (float*)(ws + bump(off, sizeof(float) * BATCH * 128 * 288));
  float* l3ws  = (float*)(ws + bump(off, sizeof(float) * BATCH * 1024));
  float* Afp3  = (float*)(ws + bump(off, sizeof(float) * BATCH * 128 * 1280));
  float* l2p   = (float*)(ws + bump(off, sizeof(float) * BATCH * 128 * 256));
  float* cat2  = (float*)(ws + bump(off, sizeof(float) * BATCH * 512 * 384));
  float* l1p   = (float*)(ws + bump(off, sizeof(float) * BATCH * 512 * 128));
  float* cat1  = (float*)(ws + bump(off, sizeof(float) * BATCH * NPTS * 160));
  float* l0p   = (float*)(ws + bump(off, sizeof(float) * BATCH * NPTS * 128));
  float* logit = (float*)(ws + bump(off, sizeof(float) * BATCH * NPTS * 50));
  bf16_t* sw_sa1   = (bf16_t*)(ws + bump(off, sizeof(bf16_t) * 8  * 1  * 512));
  bf16_t* sw_sa2   = (bf16_t*)(ws + bump(off, sizeof(bf16_t) * 16 * 5  * 512));
  bf16_t* sw_sa3   = (bf16_t*)(ws + bump(off, sizeof(bf16_t) * 64 * 9  * 512));
  bf16_t* sw_fp3   = (bf16_t*)(ws + bump(off, sizeof(bf16_t) * 16 * 40 * 512));
  bf16_t* sw_fp2   = (bf16_t*)(ws + bump(off, sizeof(bf16_t) * 8  * 12 * 512));
  bf16_t* sw_fp1   = (bf16_t*)(ws + bump(off, sizeof(bf16_t) * 8  * 5  * 512));
  bf16_t* sw_conv2 = (bf16_t*)(ws + bump(off, sizeof(bf16_t) * 4  * 4  * 512));

  auto pack = [&](const float* W, bf16_t* dst, int Kin, int Nout) {
    int KC = (Kin + 31) / 32, NT = (Nout + 15) / 16;
    int total = NT * KC * 512;
    k_pack_w<<<dim3((total + 255) / 256), dim3(256), 0, stream>>>(W, dst, Kin, Nout, KC, NT);
  };
  pack(w_sa1, sw_sa1, 6, 128);
  pack(w_sa2, sw_sa2, 131, 256);
  pack(w_sa3, sw_sa3, 259, 1024);
  pack(w_fp3, sw_fp3, 1280, 256);
  pack(w_fp2, sw_fp2, 384, 128);
  pack(w_fp1, sw_fp1, 150, 128);
  pack(w_conv2, sw_conv2, 128, 50);

  {
    int total = BATCH * NPTS * 3;
    k_transpose<<<dim3((total + 255) / 256), dim3(256), 0, stream>>>(xyz, xyzT);
  }

  // SA1
  k_fps<<<dim3(BATCH), dim3(256), 0, stream>>>(xyzT, NPTS, 512, idx1, nx1);
  {
    int total = BATCH * 512;
    k_ballquery<<<dim3((total + 255) / 256), dim3(256), 0, stream>>>(
        xyzT, nx1, NPTS, 512, 32, 0.01f, bq1, total);
  }
  k_sa1<<<dim3(512, BATCH), dim3(256), 0, stream>>>(xyzT, nx1, bq1, sw_sa1, b_sa1, l1);

  // SA2
  k_fps<<<dim3(BATCH), dim3(256), 0, stream>>>(nx1, 512, 128, idx2, nx2);
  {
    int total = BATCH * 128;
    k_ballquery<<<dim3((total + 255) / 256), dim3(256), 0, stream>>>(
        nx1, nx2, 512, 128, 64, 0.09f, bq2, total);
  }
  k_sa2<<<dim3(128, BATCH), dim3(512), 0, stream>>>(nx1, l1, nx2, bq2, sw_sa2, b_sa2, l2);

  // SA3 (global) + l3 output
  {
    int total = BATCH * 128 * 288;
    k_sa3prep<<<dim3((total + 255) / 256), dim3(256), 0, stream>>>(nx2, l2, A3);
  }
  float* out_l3 = out + (size_t)BATCH * NPTS * 50;
  k_sa3<<<dim3(64, BATCH), dim3(256), 0, stream>>>(A3, sw_sa3, b_sa3, l3ws, out_l3);

  // FP3
  {
    int total = BATCH * 128 * 1280;
    k_fp3prep<<<dim3((total + 255) / 256), dim3(256), 0, stream>>>(l2, l3ws, Afp3);
  }
  k_gemm<<<dim3(16, 8, BATCH), dim3(32), 0, stream>>>(
      Afp3, (size_t)128 * 1280, 1280, sw_fp3, 40, b_fp3,
      l2p, (size_t)128 * 256, 256, 256, 1);

  // FP2
  {
    int total = BATCH * 512 * 128;
    k_fp2prep<<<dim3((total + 255) / 256), dim3(256), 0, stream>>>(l1, cat2);
  }
  {
    int total = BATCH * 512;
    k_knn<<<dim3((total + 255) / 256), dim3(256), 0, stream>>>(
        nx1, nx2, l2p, 512, 128, 256, 256, cat2, 384, 128, total);
  }
  k_gemm<<<dim3(8, 32, BATCH), dim3(32), 0, stream>>>(
      cat2, (size_t)512 * 384, 384, sw_fp2, 12, b_fp2,
      l1p, (size_t)512 * 128, 128, 128, 1);

  // FP1
  {
    int total = BATCH * NPTS;
    k_fp1prep<<<dim3((total + 255) / 256), dim3(256), 0, stream>>>(cls, xyzT, cat1);
    k_knn<<<dim3((total + 255) / 256), dim3(256), 0, stream>>>(
        xyzT, nx1, l1p, NPTS, 512, 128, 128, cat1, 160, 22, total);
  }
  k_gemm<<<dim3(8, 256, BATCH), dim3(32), 0, stream>>>(
      cat1, (size_t)NPTS * 160, 160, sw_fp1, 5, b_fp1,
      l0p, (size_t)NPTS * 128, 128, 128, 1);

  // conv2 (logits, no relu) + log_softmax
  k_gemm<<<dim3(4, 256, BATCH), dim3(32), 0, stream>>>(
      l0p, (size_t)NPTS * 128, 128, sw_conv2, 4, b_conv2,
      logit, (size_t)NPTS * 50, 50, 50, 0);
  {
    int total = BATCH * NPTS;
    k_logsoftmax<<<dim3((total + 255) / 256), dim3(256), 0, stream>>>(logit, out, total);
  }
}
